// RNN_3590592659654
// MI455X (gfx1250) — compile-verified, matrix operand order checked by reading, not run on hardware
//
#include <hip/hip_runtime.h>

// ---------------------------------------------------------------------------
// Shapes (compile-time constants from the reference)
// ---------------------------------------------------------------------------
#define T_DIM 512
#define B_DIM 64
#define K_DIM 512      // D_IN
#define N_DIM 1024     // H
#define M_DIM (T_DIM * B_DIM)   // 32768 flattened rows

#define MBLK 128
#define NBLK 128
#define KSTEP 32
#define NSTEPS (K_DIM / KSTEP)  // 16
#define LDS_ROW 20              // padded A/B row stride in dwords (80B, 16B aligned)
#define SP_ROW 132              // padded sp row stride in floats (528B, 16B aligned)

typedef __attribute__((ext_vector_type(16))) __bf16       v16bf;
typedef __attribute__((ext_vector_type(2)))  __bf16       v2bf;
typedef __attribute__((ext_vector_type(8)))  float        v8f;
typedef __attribute__((ext_vector_type(4)))  unsigned int u32x4;

union FragU {
    u32x4 u[2];   // two 16B LDS loads
    v16bf bf;     // WMMA operand view
};

// f32 pair -> packed bf16 dword
__device__ __forceinline__ unsigned int pack_bf16(float lo, float hi) {
#if __has_builtin(__builtin_amdgcn_cvt_pk_bf16_f32)
    union { v2bf v; unsigned int u; } cv;
    cv.v = __builtin_amdgcn_cvt_pk_bf16_f32(lo, hi);
    return cv.u;
#else
    // hope for hardware fptrunc legalization (v_cvt_pk_bf16_f32 on CDNA4+)
    union { v2bf v; unsigned int u; } cv;
    cv.v = (v2bf){(__bf16)lo, (__bf16)hi};
    return cv.u;
#endif
}

// tanh: prefer hardware V_TANH_F32 (listed as a TRANS32 op in the CDNA5 ISA);
// fallback: branch-free 1 - 2/(exp2(2*log2e*x)+1), saturates to +-1.
__device__ __forceinline__ float fast_tanh(float x) {
#if __has_builtin(__builtin_amdgcn_tanhf)
    return __builtin_amdgcn_tanhf(x);
#else
    float e = __builtin_amdgcn_exp2f(x * 2.8853900817779268f); // e^(2x)
    return 1.0f - 2.0f * __builtin_amdgcn_rcpf(e + 1.0f);
#endif
}

// ---------------------------------------------------------------------------
// Kernel 1: state_proj[b,h] = state[b,:] @ W_h[:,h] + b_h[h]
//           also copies `state` to the tail of d_out (second tuple output).
// grid = (N_DIM/256, B_DIM), block = 256
// ---------------------------------------------------------------------------
__global__ __launch_bounds__(256)
void rnn_state_proj(const float* __restrict__ state,
                    const float* __restrict__ W_h,
                    const float* __restrict__ b_h,
                    float* __restrict__ sp,
                    float* __restrict__ out_tail) {
    __shared__ float srow_lds[N_DIM];
    const int tid = threadIdx.x;
    const int h = blockIdx.x * 256 + tid;
    const int b = blockIdx.y;
    const float* srow = state + (size_t)b * N_DIM;

    #pragma unroll
    for (int k = tid; k < N_DIM; k += 256) srow_lds[k] = srow[k];
    __syncthreads();

    float acc = b_h[h];
    #pragma unroll 8
    for (int k = 0; k < N_DIM; ++k)
        acc = fmaf(srow_lds[k], W_h[(size_t)k * N_DIM + h], acc);

    sp[(size_t)b * N_DIM + h] = acc;
    out_tail[(size_t)b * N_DIM + h] = srow[h];
}

// ---------------------------------------------------------------------------
// Kernel 2: out[m,h] = tanh( X[m,:] @ W_x[:,h] + sp[m % 64, h] )
// grid = (M_DIM/MBLK, N_DIM/NBLK), block = 256 (8 wave32 waves, 2x4 grid)
// Double-buffered LDS, bf16 WMMA, LDS-cached state_proj tile, NT stores.
// ---------------------------------------------------------------------------
__global__ __launch_bounds__(256)
void rnn_gemm_bf16(const float* __restrict__ X,
                   const float* __restrict__ Wx,
                   const float* __restrict__ sp,
                   float* __restrict__ out) {
    __shared__ __align__(16) unsigned int ldsA[2][MBLK * LDS_ROW];
    __shared__ __align__(16) unsigned int ldsB[2][NBLK * LDS_ROW];
    __shared__ __align__(16) float        ldsSP[B_DIM * SP_ROW];

    const int tid    = threadIdx.x;
    const int lane   = tid & 31;
    const int wid    = tid >> 5;
    const int wm     = wid & 1;        // wave M position (0..1) -> 64 rows
    const int wn     = wid >> 1;       // wave N position (0..3) -> 32 cols
    const int laneLo = lane & 15;
    const int laneHi = lane >> 4;      // K-half select per ISA fragment layout

    const int mBlock = blockIdx.x * MBLK;
    const int nBlock = blockIdx.y * NBLK;

    // loader roles
    const int xr  = tid >> 1;          // 0..127 : X tile row
    const int xh  = tid & 1;           // 16-wide K half
    const int wcn = tid & 127;         // 0..127 : W tile column (N)
    const int wkg = tid >> 7;          // 0..1   : K group of 16

    // ---- stage the state_proj tile (64 x 128) into LDS: shared by all M-blocks
    {
        const int c4 = (tid & 31) * 4;     // column, step 4
        const int r0 = tid >> 5;           // base row 0..7
        #pragma unroll
        for (int rr = 0; rr < 8; ++rr) {
            const int r = r0 + rr * 8;     // 0..63
            float4 v = *(const float4*)&sp[(size_t)r * N_DIM + nBlock + c4];
            *(float4*)&ldsSP[r * SP_ROW + c4] = v;
        }
    }

    v8f c[4][2];
    #pragma unroll
    for (int i = 0; i < 4; ++i)
        #pragma unroll
        for (int j = 0; j < 2; ++j)
            c[i][j] = (v8f){0.f, 0.f, 0.f, 0.f, 0.f, 0.f, 0.f, 0.f};

    // global prefetch registers
    float4 xa0, xa1, xa2, xa3;
    float  wv[16];

    const float* Xbase = X + (size_t)(mBlock + xr) * K_DIM + xh * 16;
    const float* Wbase = Wx + (size_t)(wkg * 16) * N_DIM + nBlock + wcn;

    auto loadGlobal = [&](int kk) {
        const float* src = Xbase + kk;
        xa0 = *(const float4*)(src + 0);
        xa1 = *(const float4*)(src + 4);
        xa2 = *(const float4*)(src + 8);
        xa3 = *(const float4*)(src + 12);
        const float* wsrc = Wbase + (size_t)kk * N_DIM;
        #pragma unroll
        for (int k = 0; k < 16; ++k) wv[k] = wsrc[(size_t)k * N_DIM];
    };

    auto storeLds = [&](int p) {
        u32x4 pa0 = { pack_bf16(xa0.x, xa0.y), pack_bf16(xa0.z, xa0.w),
                      pack_bf16(xa1.x, xa1.y), pack_bf16(xa1.z, xa1.w) };
        u32x4 pa1 = { pack_bf16(xa2.x, xa2.y), pack_bf16(xa2.z, xa2.w),
                      pack_bf16(xa3.x, xa3.y), pack_bf16(xa3.z, xa3.w) };
        unsigned int* dA = &ldsA[p][xr * LDS_ROW + xh * 8];
        *(u32x4*)(dA + 0) = pa0;
        *(u32x4*)(dA + 4) = pa1;
        u32x4 pb0 = { pack_bf16(wv[0],  wv[1]),  pack_bf16(wv[2],  wv[3]),
                      pack_bf16(wv[4],  wv[5]),  pack_bf16(wv[6],  wv[7]) };
        u32x4 pb1 = { pack_bf16(wv[8],  wv[9]),  pack_bf16(wv[10], wv[11]),
                      pack_bf16(wv[12], wv[13]), pack_bf16(wv[14], wv[15]) };
        unsigned int* dB = &ldsB[p][wcn * LDS_ROW + wkg * 8];
        *(u32x4*)(dB + 0) = pb0;
        *(u32x4*)(dB + 4) = pb1;
    };

    loadGlobal(0);
    storeLds(0);
    __syncthreads();

    int p = 0;
    for (int step = 0; step < NSTEPS; ++step) {
        if (step + 1 < NSTEPS) loadGlobal((step + 1) * KSTEP);  // overlap with WMMA

        // fragments per ISA 16-bit layout: two b128 loads each
        const unsigned int* bA = ldsA[p];
        const unsigned int* bB = ldsB[p];
        FragU a[4];
        #pragma unroll
        for (int i = 0; i < 4; ++i) {
            const int row = wm * 64 + i * 16 + laneLo;
            a[i].u[0] = *(const u32x4*)&bA[row * LDS_ROW + laneHi * 4];      // K 0..7 / 8..15
            a[i].u[1] = *(const u32x4*)&bA[row * LDS_ROW + 8 + laneHi * 4];  // K 16..23 / 24..31
        }
        FragU bfr[2];
        #pragma unroll
        for (int j = 0; j < 2; ++j) {
            const int col = wn * 32 + j * 16 + laneLo;
            bfr[j].u[0] = *(const u32x4*)&bB[col * LDS_ROW + laneHi * 4];
            bfr[j].u[1] = *(const u32x4*)&bB[col * LDS_ROW + 8 + laneHi * 4];
        }

        #pragma unroll
        for (int i = 0; i < 4; ++i)
            #pragma unroll
            for (int j = 0; j < 2; ++j)
                c[i][j] = __builtin_amdgcn_wmma_f32_16x16x32_bf16(
                    false, a[i].bf, false, bfr[j].bf,
                    (short)0, c[i][j], false, false);

        if (step + 1 < NSTEPS) {
            storeLds(p ^ 1);   // write other buffer (no conflict with readers)
            __syncthreads();   // one barrier per K-step
            p ^= 1;
        }
    }

    // ---- epilogue: add state_proj from LDS, hw tanh, NT streaming stores ----
    float* outBase = out + (size_t)(mBlock + wm * 64 + laneHi * 8) * N_DIM
                         + nBlock + wn * 32 + laneLo;
    const float* spBase = &ldsSP[(laneHi * 8) * SP_ROW + wn * 32 + laneLo];

    #pragma unroll
    for (int i = 0; i < 4; ++i) {
        #pragma unroll
        for (int j = 0; j < 2; ++j) {
            #pragma unroll
            for (int v = 0; v < 8; ++v) {
                const int mo = i * 16 + v;                 // b = mo + laneHi*8 (in [0,63])
                const float s = spBase[mo * SP_ROW + j * 16];
                const float r = fast_tanh(c[i][j][v] + s);
                __builtin_nontemporal_store(r, &outBase[(size_t)mo * N_DIM + j * 16]);
            }
        }
    }
}

// ---------------------------------------------------------------------------
// Host-side launch
// ---------------------------------------------------------------------------
extern "C" void kernel_launch(void* const* d_in, const int* in_sizes, int n_in,
                              void* d_out, int out_size, void* d_ws, size_t ws_size,
                              hipStream_t stream) {
    const float* X     = (const float*)d_in[0];   // [T,B,D_IN]
    const float* state = (const float*)d_in[1];   // [B,H]
    const float* W_x   = (const float*)d_in[2];   // [D_IN,H]
    const float* W_h   = (const float*)d_in[3];   // [H,H]
    const float* b_h   = (const float*)d_in[4];   // [H]

    float* out      = (float*)d_out;                       // [T,B,H] then [B,H]
    float* out_tail = out + (size_t)T_DIM * B_DIM * N_DIM; // state passthrough
    float* sp       = (float*)d_ws;                        // [B,H] = 256 KB scratch

    rnn_state_proj<<<dim3(N_DIM / 256, B_DIM), 256, 0, stream>>>(
        state, W_h, b_h, sp, out_tail);

    rnn_gemm_bf16<<<dim3(M_DIM / MBLK, N_DIM / NBLK), 256, 0, stream>>>(
        X, W_x, sp, out);
}
